// DGP_RF_Embeddings_87857851007243
// MI455X (gfx1250) — compile-verified, matrix operand order checked by reading, not run on hardware
//
#include <hip/hip_runtime.h>

#define N_PTS  131072
#define SEGS   16384
#define DIN    256
#define NRF    1024
#define DOUT   128
#define SD     (SEGS * DOUT)          // 2097152 floats per output half

typedef __attribute__((ext_vector_type(16))) __bf16 v16bf;
typedef __attribute__((ext_vector_type(8)))  float  v8f;
typedef __attribute__((ext_vector_type(4)))  float  v4f;

// ws layout (bf16 element offsets). Each tile = 32 lanes * 16 bf16 = 512 elems.
#define WS_W1MU   0         // 512 tiles  (64 rf-tiles x 8 k-chunks)
#define WS_W1VAR  262144    // 512 tiles
#define WS_W2MU   524288    // 256 tiles  (8 dout-tiles x 32 rf-chunks)
#define WS_W2MUSQ 655360    // 256 tiles
#define WS_W2VAR  786432    // 256 tiles
// total 917504 bf16 = 1.75 MB of d_ws

// LDS layout (bf16 elements), 2 waves per block:
//   [0 .. 16384)      : per-wave X staging (8192 each: 4096 x, 4096 x^2)
//   [16384 .. 28672)  : GEMM2 weight stage, 24 tiles * 512 (si = m*8 + dt)
#define LDS_G2 16384
#define LDS_ELEMS 28672                // 57344 bytes

__device__ __forceinline__ v8f wmma_bf16(v16bf a, v16bf b, v8f c) {
    return __builtin_amdgcn_wmma_f32_16x16x32_bf16(false, a, false, b,
                                                   (short)0, c, false, false);
}

__device__ __forceinline__ void atomic_add_f32(float* p, float v) {
    // no-return fp32 global atomic add (STOREcnt; s_endpgm's implicit wait covers it)
    asm volatile("global_atomic_add_f32 %0, %1, off" : : "v"(p), "v"(v) : "memory");
}

// CDNA5 async copy: 16 bytes per lane, global -> LDS, tracked by ASYNCcnt.
__device__ __forceinline__ void async_g2l_b128(unsigned lds_off, const void* gaddr) {
    asm volatile("global_load_async_to_lds_b128 %0, %1, off"
                 : : "v"(lds_off), "v"(gaddr) : "memory");
}

__device__ __forceinline__ void wait_asynccnt0() {
    asm volatile("s_wait_asynccnt 0x0" ::: "memory");
}

__device__ __forceinline__ void relu_moments(float mu, float va, float& mo, float& vo) {
    float vps = va + 1e-8f;
    float rs  = rsqrtf(vps);
    float s   = vps * rs;                 // sqrt(va + eps)
    float z   = mu * rs;
    float Phi = 0.5f * (1.0f + erff(z * 0.70710678118654752f));
    float phi = __expf(-0.5f * z * z) * 0.39894228040143268f;
    mo = mu * Phi + s * phi;
    float ey2 = (mu * mu + va) * Phi + mu * s * phi;
    vo = fmaxf(ey2 - mo * mo, 1e-6f);
}

// ---------------------------------------------------------------------------
__global__ __launch_bounds__(256) void zero_out_kernel(float* __restrict__ out) {
    size_t i = ((size_t)blockIdx.x * 256 + threadIdx.x) * 4;
    v4f z = {0.f, 0.f, 0.f, 0.f};
    *(v4f*)(out + i) = z;
}

// ---------------------------------------------------------------------------
// Convert weights to bf16, pre-swizzled into WMMA A-tile per-lane layout:
// lane L<16 : row m0+L,    K cols {k0..k0+7, k0+16..k0+23}
// lane L>=16: row m0+L-16, K cols {k0+8..k0+15, k0+24..k0+31}
__global__ __launch_bounds__(256) void prep_weights(
    const float* __restrict__ w1mu, const float* __restrict__ w1var,
    const float* __restrict__ w2mu, const float* __restrict__ w2var,
    __bf16* __restrict__ ws)
{
    int u    = (blockIdx.x * 256 + threadIdx.x) >> 5;   // tile unit id, [0,1792)
    int lane = threadIdx.x & 31;
    int hf   = lane >> 4;
    int r    = lane & 15;

    const float* src;
    __bf16* dst;
    int row, colbase, rowstride;
    bool square = false;

    if (u < 512)       { int t = u;        src = w1mu;  dst = ws + WS_W1MU   + (size_t)t * 512;
                         row = (t >> 3) * 16 + r; colbase = (t & 7)  * 32 + hf * 8; rowstride = DIN; }
    else if (u < 1024) { int t = u - 512;  src = w1var; dst = ws + WS_W1VAR  + (size_t)t * 512;
                         row = (t >> 3) * 16 + r; colbase = (t & 7)  * 32 + hf * 8; rowstride = DIN; }
    else if (u < 1280) { int t = u - 1024; src = w2mu;  dst = ws + WS_W2MU   + (size_t)t * 512;
                         row = (t >> 5) * 16 + r; colbase = (t & 31) * 32 + hf * 8; rowstride = NRF; }
    else if (u < 1536) { int t = u - 1280; src = w2mu;  dst = ws + WS_W2MUSQ + (size_t)t * 512;
                         row = (t >> 5) * 16 + r; colbase = (t & 31) * 32 + hf * 8; rowstride = NRF;
                         square = true; }
    else               { int t = u - 1536; src = w2var; dst = ws + WS_W2VAR  + (size_t)t * 512;
                         row = (t >> 5) * 16 + r; colbase = (t & 31) * 32 + hf * 8; rowstride = NRF; }

    const float* p0 = src + (size_t)row * rowstride + colbase;
    v4f a0 = *(const v4f*)(p0);
    v4f a1 = *(const v4f*)(p0 + 4);
    v4f b0 = *(const v4f*)(p0 + 16);
    v4f b1 = *(const v4f*)(p0 + 20);
    __bf16* d = dst + lane * 16;
#pragma unroll
    for (int i = 0; i < 4; ++i) {
        float x0 = a0[i], x1 = a1[i], x2 = b0[i], x3 = b1[i];
        if (square) { x0 *= x0; x1 *= x1; x2 *= x2; x3 *= x3; }
        d[i]      = (__bf16)x0;
        d[4 + i]  = (__bf16)x1;
        d[8 + i]  = (__bf16)x2;
        d[12 + i] = (__bf16)x3;
    }
}

// ---------------------------------------------------------------------------
// Fused DGP-RF. One wave = 16 points. 2 waves/block share LDS-staged GEMM2
// weight tiles (async-copied) each RF chunk; GEMM1's D registers feed GEMM2's
// B operand directly (same lane layout) after packed bf16 conversion.
__global__ __launch_bounds__(64) void dgp_fused(
    const float* __restrict__ X, const int* __restrict__ Xidx,
    const __bf16* __restrict__ ws, float* __restrict__ out)
{
    __shared__ __bf16 lds[LDS_ELEMS];

    const int lane = threadIdx.x & 31;
    const int wib  = threadIdx.x >> 5;            // 0..1
    const int wave = blockIdx.x * 2 + wib;
    const int hf   = lane >> 4;
    const int p    = wave * 16 + (lane & 15);

    const float* xrow = X + (size_t)p * DIN;
    __bf16* lbase = &lds[wib * 8192];

    // Stage this wave's X tile as bf16(x) and bf16(x^2) in B-matrix layout.
#pragma unroll
    for (int kk = 0; kk < 8; ++kk) {
        int c0 = kk * 32 + hf * 8;
        v4f a0 = *(const v4f*)(xrow + c0);
        v4f a1 = *(const v4f*)(xrow + c0 + 4);
        v4f b0 = *(const v4f*)(xrow + c0 + 16);
        v4f b1 = *(const v4f*)(xrow + c0 + 20);
        __bf16* dx = lbase + kk * 512 + lane * 16;
        __bf16* dq = dx + 4096;
#pragma unroll
        for (int i = 0; i < 4; ++i) {
            dx[i]      = (__bf16)a0[i];
            dx[4 + i]  = (__bf16)a1[i];
            dx[8 + i]  = (__bf16)b0[i];
            dx[12 + i] = (__bf16)b1[i];
            dq[i]      = (__bf16)(a0[i] * a0[i]);
            dq[4 + i]  = (__bf16)(a1[i] * a1[i]);
            dq[8 + i]  = (__bf16)(b0[i] * b0[i]);
            dq[12 + i] = (__bf16)(b1[i] * b1[i]);
        }
    }
    // X regions are wave-private, DS ops are in-order per wave: no barrier yet.

    v8f zf = {0.f, 0.f, 0.f, 0.f, 0.f, 0.f, 0.f, 0.f};
    v8f accM[8], accV[8];
#pragma unroll
    for (int t = 0; t < 8; ++t) { accM[t] = zf; accV[t] = zf; }

    const size_t lo = (size_t)lane * 16;
    const unsigned g2base = (unsigned)(size_t)&lds[LDS_G2];

    __syncthreads();   // G2 stage region free for iteration 0

#pragma unroll 1
    for (int rc = 0; rc < 32; ++rc) {     // 32 RF chunks of 32
        // --- async-stage this chunk's 24 GEMM2 tiles (no data VGPRs used) ---
        // si = m*8 + dt ; m: 0=W2mu 1=W2mu^2 2=W2var. Wave wib copies 12 tiles.
        {
            const __bf16* w2base[3] = { ws + WS_W2MU, ws + WS_W2MUSQ, ws + WS_W2VAR };
#pragma unroll
            for (int s = 0; s < 12; ++s) {
                int si = wib * 12 + s;
                int m  = si >> 3, dt = si & 7;
                const __bf16* gA = w2base[m] + (size_t)(dt * 32 + rc) * 512 + lane * 8;
                unsigned lA = g2base + (unsigned)si * 1024u + (unsigned)lane * 16u;
                async_g2l_b128(lA,        gA);         // bytes [0,512) of tile
                async_g2l_b128(lA + 512u, gA + 256);   // bytes [512,1024)
            }
        }

        // --- GEMM1 (overlaps the async copies) ---
        v8f m0 = zf, m1 = zf, q0 = zf, q1 = zf;
#pragma unroll
        for (int kk = 0; kk < 8; ++kk) {  // D_IN in K=32 steps
            v16bf bx  = *(const v16bf*)(lbase + kk * 512 + lane * 16);
            v16bf bxs = *(const v16bf*)(lbase + 4096 + kk * 512 + lane * 16);
            size_t t0 = (size_t)((rc * 2    ) * 8 + kk) * 512 + lo;
            size_t t1 = (size_t)((rc * 2 + 1) * 8 + kk) * 512 + lo;
            v16bf am0 = *(const v16bf*)(ws + WS_W1MU  + t0);
            v16bf am1 = *(const v16bf*)(ws + WS_W1MU  + t1);
            v16bf av0 = *(const v16bf*)(ws + WS_W1VAR + t0);
            v16bf av1 = *(const v16bf*)(ws + WS_W1VAR + t1);
            m0 = wmma_bf16(am0, bx,  m0);
            m1 = wmma_bf16(am1, bx,  m1);
            q0 = wmma_bf16(av0, bxs, q0);
            q1 = wmma_bf16(av1, bxs, q1);
        }

        // --- ReLU moments on D registers; pack GEMM2 B-matrices ---
        v16bf bm, bv, bmv;
#pragma unroll
        for (int i = 0; i < 8; ++i) {
            float mo, vo;
            relu_moments(m0[i], q0[i], mo, vo);
            bm[i]      = (__bf16)mo;
            bv[i]      = (__bf16)vo;
            bmv[i]     = (__bf16)(mo * mo + vo);
            relu_moments(m1[i], q1[i], mo, vo);
            bm[8 + i]  = (__bf16)mo;
            bv[8 + i]  = (__bf16)vo;
            bmv[8 + i] = (__bf16)(mo * mo + vo);
        }

        // --- make staged tiles visible to both waves ---
        wait_asynccnt0();
        __syncthreads();

        // --- GEMM2 from LDS, 1-deep software pipeline on the A tiles so each
        //     WMMA's operands are already resident (wait only for dt+1 loads) ---
        {
            const __bf16* g2p = &lds[LDS_G2 + lo];
            v16bf a1 = *(const v16bf*)(g2p);           // si = 0*8+dt : W2mu
            v16bf a2 = *(const v16bf*)(g2p + 4096);    // si = 1*8+dt : W2mu^2
            v16bf a3 = *(const v16bf*)(g2p + 8192);    // si = 2*8+dt : W2var
#pragma unroll
            for (int dt = 0; dt < 8; ++dt) {
                v16bf n1 = a1, n2 = a2, n3 = a3;
                if (dt < 7) {
                    const __bf16* gn = &lds[LDS_G2 + (size_t)(dt + 1) * 512 + lo];
                    n1 = *(const v16bf*)(gn);
                    n2 = *(const v16bf*)(gn + 4096);
                    n3 = *(const v16bf*)(gn + 8192);
                }
                accM[dt] = wmma_bf16(a1, bm,  accM[dt]);
                accV[dt] = wmma_bf16(a2, bv,  accV[dt]);
                accV[dt] = wmma_bf16(a3, bmv, accV[dt]);
                a1 = n1; a2 = n2; a3 = n3;
            }
        }

        __syncthreads();   // all waves done reading before next chunk overwrites
    }

    // Precision-weighted segment accumulation.
    // D layout: lane holds point p = lane&15, douts dt*16 + hf*8 + g.
    int seg = Xidx[p];
    float* pmAcc = out + (size_t)seg * DOUT;        // sum(prec * mean) region
    float* prAcc = out + SD + (size_t)seg * DOUT;   // sum(prec) region
#pragma unroll
    for (int dt = 0; dt < 8; ++dt) {
#pragma unroll
        for (int g = 0; g < 8; ++g) {
            int dout = dt * 16 + hf * 8 + g;
            float prec = 1.0f / accV[dt][g];
            atomic_add_f32(prAcc + dout, prec);
            atomic_add_f32(pmAcc + dout, prec * accM[dt][g]);
        }
    }
}

// ---------------------------------------------------------------------------
// embedd_vars = 1/sum(prec); embedd_means = sum(prec*mean) * embedd_vars
__global__ __launch_bounds__(256) void finalize_kernel(float* __restrict__ out) {
    size_t i = ((size_t)blockIdx.x * 256 + threadIdx.x) * 4;
    v4f pm = *(v4f*)(out + i);
    v4f pr = *(v4f*)(out + SD + i);
    v4f var, mean;
#pragma unroll
    for (int j = 0; j < 4; ++j) {
        var[j]  = 1.0f / pr[j];
        mean[j] = pm[j] * var[j];
    }
    *(v4f*)(out + i)      = mean;
    *(v4f*)(out + SD + i) = var;
}

// ---------------------------------------------------------------------------
extern "C" void kernel_launch(void* const* d_in, const int* in_sizes, int n_in,
                              void* d_out, int out_size, void* d_ws, size_t ws_size,
                              hipStream_t stream) {
    const float* X     = (const float*)d_in[0];
    const int*   Xidx  = (const int*)  d_in[1];
    const float* w1mu  = (const float*)d_in[2];
    const float* w1var = (const float*)d_in[3];
    const float* w2mu  = (const float*)d_in[4];
    const float* w2var = (const float*)d_in[5];
    float*  out = (float*)d_out;
    __bf16* ws  = (__bf16*)d_ws;

    zero_out_kernel<<<4096, 256, 0, stream>>>(out);                        // 2*SD floats
    prep_weights  <<<224, 256, 0, stream>>>(w1mu, w1var, w2mu, w2var, ws); // 1792 tile-waves
    dgp_fused     <<<4096, 64, 0, stream>>>(X, Xidx, ws, out);             // 8192 waves * 16 pts
    finalize_kernel<<<2048, 256, 0, stream>>>(out);                        // SD floats
}
